// InfoNCELoss_84284438217273
// MI455X (gfx1250) — matrix operand order repaired and checked by reading, not compiled, not run on hardware
//
#include <hip/hip_runtime.h>

typedef __attribute__((ext_vector_type(16))) _Float16 v16h;
typedef __attribute__((ext_vector_type(4)))  _Float16 v4h;
typedef __attribute__((ext_vector_type(8)))  float    v8f;

#define BSZ   8192
#define DD    128
// exp(dot * 10) == exp2(dot * 10 * log2(e))
#define SCALE_EXP2 14.4269504088896340736f

// ---------------------------------------------------------------------------
// Kernel 1: L2-normalize rows, emit f16 row-major [BSZ][DD].
// One wave (32 lanes) per row; 4 floats per lane.
// ---------------------------------------------------------------------------
__global__ __launch_bounds__(256) void k_norm(const float* __restrict__ emb,
                                              _Float16* __restrict__ out) {
  const int wave = threadIdx.x >> 5;
  const int lane = threadIdx.x & 31;
  const int row  = blockIdx.x * 8 + wave;

  const float4 v = ((const float4*)(emb + (size_t)row * DD))[lane];
  float ss = v.x * v.x + v.y * v.y + v.z * v.z + v.w * v.w;
#pragma unroll
  for (int off = 16; off > 0; off >>= 1) ss += __shfl_xor(ss, off, 32);

  const float nrm = fmaxf(sqrtf(ss), 1e-12f);
  const float inv = 1.0f / nrm;

  v4h h;
  h.x = (_Float16)(v.x * inv);
  h.y = (_Float16)(v.y * inv);
  h.z = (_Float16)(v.z * inv);
  h.w = (_Float16)(v.w * inv);
  *(v4h*)(out + (size_t)row * DD + lane * 4) = h;
}

// ---------------------------------------------------------------------------
// Kernel 2: zero the pos/all accumulators.
// ---------------------------------------------------------------------------
__global__ __launch_bounds__(256) void k_zero(float* __restrict__ p, int n) {
  const int i = blockIdx.x * 256 + threadIdx.x;
  if (i < n) p[i] = 0.0f;
}

// ---------------------------------------------------------------------------
// Kernel 3: fused  exp(E_n . E_n^T / T)  masked row-sum.
// Grid: (BSZ/128, 16).  Block: 256 threads = 8 waves.
// Wave w owns rows [blk*128 + 16w, +16); block column-chunk = 512 cols.
// A operand (16x128 f16) lives in VGPRs; B tiles double-buffered in LDS so
// the next tile's global loads overlap the current tile's WMMA+exp epilogue
// (single barrier per iteration). The j==i diagonal mask is only evaluated
// in the single tile per wave where the row/col ranges coincide.
// ---------------------------------------------------------------------------
__global__ __launch_bounds__(256) void k_sim(const _Float16* __restrict__ E,
                                             const int* __restrict__ labels,
                                             float* __restrict__ pos_sum,
                                             float* __restrict__ all_sum) {
  __shared__ uint4 lds[2][16][17];   // padded: bank-conflict-free ds_load_b128
  __shared__ int   lab_lds[2][16];

  const int tid  = threadIdx.x;
  const int wave = tid >> 5;
  const int lane = tid & 31;
  const int n    = lane & 15;   // N index within 16-wide tile
  const int g    = lane >> 4;   // lane half-group

  const int ibase = blockIdx.x * 128 + wave * 16;

  // ---- A operand: 4 chunks of 16x32 f16 (ISA 7.12.2 A layout) -------------
  union AU { uint4 u[2]; v16h v; };
  AU a[4];
  const uint4* arow = (const uint4*)(E + (size_t)(ibase + n) * DD);
#pragma unroll
  for (int c = 0; c < 4; ++c) {
    a[c].u[0] = arow[4 * c + g];       // K = 32c + 8g      .. +7
    a[c].u[1] = arow[4 * c + 2 + g];   // K = 32c + 16 + 8g .. +7
  }

  int lab_i[8];
#pragma unroll
  for (int r = 0; r < 8; ++r) lab_i[r] = labels[ibase + 8 * g + r];

  float pos_acc[8], all_acc[8];
#pragma unroll
  for (int r = 0; r < 8; ++r) { pos_acc[r] = 0.0f; all_acc[r] = 0.0f; }

  const int jt0 = blockIdx.y * 32;    // 32 column tiles of 16 per chunk
  const int rr  = tid >> 4;           // cooperative fill: 16 rows x 16 uint4
  const int cc  = tid & 15;

  // prologue: stage tile 0
  {
    const int jb = jt0 * 16;
    lds[0][rr][cc] = ((const uint4*)(E + (size_t)(jb + rr) * DD))[cc];
    if (tid < 16) lab_lds[0][tid] = labels[jb + tid];
  }

  for (int t = 0; t < 32; ++t) {
    const int buf = t & 1;
    __syncthreads();   // stage(t) visible; stage buffer (t+1)&1 free again

    if (t + 1 < 32) {  // uniform branch: prefetch next tile into other buffer
      const int jb = (jt0 + t + 1) * 16;
      lds[buf ^ 1][rr][cc] = ((const uint4*)(E + (size_t)(jb + rr) * DD))[cc];
      if (tid < 16) lab_lds[buf ^ 1][tid] = labels[jb + tid];
    }

    // ---- 16x16 tile of sim via 4 chained f16 WMMAs (K = 128) --------------
    v8f acc = {};
#pragma unroll
    for (int c = 0; c < 4; ++c) {
      union BU { uint4 u[2]; v16h v; } b;
      // B 32x16 layout: same K order as A (K = 8g..+7, then 16+8g..+7)
      b.u[0] = lds[buf][n][4 * c + g];
      b.u[1] = lds[buf][n][4 * c + 2 + g];
      acc = __builtin_amdgcn_wmma_f32_16x16x32_f16(
          false, a[c].v, false, b.v, (short)0, acc, false, false);
    }

    const int jbase = (jt0 + t) * 16;
    const int lab_j = lab_lds[buf][n];

    if (jbase != ibase) {
      // common case (511/512 tiles): no diagonal element in this tile
#pragma unroll
      for (int r = 0; r < 8; ++r) {
        const float e = __builtin_amdgcn_exp2f(acc[r] * SCALE_EXP2);
        all_acc[r] += e;
        pos_acc[r] += (lab_j == lab_i[r]) ? e : 0.0f;
      }
    } else {
      // diagonal tile: exclude j == i
      const int j = jbase + n;
#pragma unroll
      for (int r = 0; r < 8; ++r) {
        const int  i       = ibase + 8 * g + r;
        const float e      = __builtin_amdgcn_exp2f(acc[r] * SCALE_EXP2);
        const bool offdiag = (j != i);
        all_acc[r] += offdiag ? e : 0.0f;
        pos_acc[r] += (offdiag && (lab_j == lab_i[r])) ? e : 0.0f;
      }
    }
  }

  // ---- reduce over the 16 N-lanes of each half-group, then atomics --------
#pragma unroll
  for (int r = 0; r < 8; ++r) {
    float p = pos_acc[r], al = all_acc[r];
#pragma unroll
    for (int off = 1; off < 16; off <<= 1) {
      p  += __shfl_xor(p,  off, 32);
      al += __shfl_xor(al, off, 32);
    }
    if (n == 0) {
      atomicAdd(&pos_sum[ibase + 8 * g + r], p);
      atomicAdd(&all_sum[ibase + 8 * g + r], al);
    }
  }
}

// ---------------------------------------------------------------------------
// Kernel 4: final scalar loss = mean over valid rows of -log(pos/all).
// ---------------------------------------------------------------------------
__global__ __launch_bounds__(256) void k_loss(const float* __restrict__ pos_sum,
                                              const float* __restrict__ all_sum,
                                              float* __restrict__ out) {
  __shared__ float s_loss[256];
  __shared__ float s_cnt[256];
  float L = 0.0f, C = 0.0f;
  for (int i = threadIdx.x; i < BSZ; i += 256) {
    const float p = pos_sum[i];
    const float a = all_sum[i];
    if (p > 0.0f) { L += logf(a) - logf(p); C += 1.0f; }
  }
  s_loss[threadIdx.x] = L;
  s_cnt[threadIdx.x]  = C;
  __syncthreads();
  for (int s = 128; s > 0; s >>= 1) {
    if (threadIdx.x < s) {
      s_loss[threadIdx.x] += s_loss[threadIdx.x + s];
      s_cnt[threadIdx.x]  += s_cnt[threadIdx.x + s];
    }
    __syncthreads();
  }
  if (threadIdx.x == 0) out[0] = s_loss[0] / fmaxf(s_cnt[0], 1.0f);
}

// ---------------------------------------------------------------------------
extern "C" void kernel_launch(void* const* d_in, const int* in_sizes, int n_in,
                              void* d_out, int out_size, void* d_ws, size_t ws_size,
                              hipStream_t stream) {
  const float* emb    = (const float*)d_in[0];
  const int*   labels = (const int*)d_in[1];
  float*       out    = (float*)d_out;

  char* ws = (char*)d_ws;
  _Float16* normF16 = (_Float16*)ws;                         // 8192*128*2 = 2 MB
  float* pos_sum = (float*)(ws + (size_t)BSZ * DD * 2);      // 32 KB
  float* all_sum = pos_sum + BSZ;                            // 32 KB

  k_norm<<<BSZ / 8, 256, 0, stream>>>(emb, normF16);
  k_zero<<<(2 * BSZ) / 256, 256, 0, stream>>>(pos_sum, 2 * BSZ);
  k_sim<<<dim3(BSZ / 128, 16), 256, 0, stream>>>(normF16, labels, pos_sum, all_sum);
  k_loss<<<1, 256, 0, stream>>>(pos_sum, all_sum, out);
}
